// MaskedSelfAttention_85521388798145
// MI455X (gfx1250) — matrix operand order repaired and checked by reading, not compiled, not run on hardware
//
#include <hip/hip_runtime.h>
#include <hip/hip_bf16.h>

typedef __attribute__((ext_vector_type(16))) __bf16 v16bf;
typedef __attribute__((ext_vector_type(8)))  __bf16 v8bf;
typedef __attribute__((ext_vector_type(8)))  float  v8f;
typedef __attribute__((ext_vector_type(4)))  unsigned int v4u;
typedef __attribute__((ext_vector_type(8)))  int    v8i;
typedef __attribute__((ext_vector_type(4)))  int    v4i;

#define B_   2
#define S_   2048
#define D_   1024
#define H_   16
#define DK_  64
#define MTOT (B_*S_)          // 4096 rows

// ---------------------------------------------------------------------------
// Tensor Data Mover: build a D# (ISA 8.3/8.4) for a 2D tile of bf16 (2-byte)
// elements and issue TENSOR_LOAD_TO_LDS. Group0: count=1/type=2/lds/global
// addr. Group1: data_size=1 (2B), tensor_dim0/1, tile_dim0/1, dim0 stride.
// Completion via TENSORcnt -> s_wait_tensorcnt 0.
// ---------------------------------------------------------------------------
__device__ __forceinline__ void tdm_load_2d_bf16(void* lds_dst, const void* gsrc,
                                                 unsigned tile_d0, unsigned tile_d1,
                                                 unsigned tensor_d0, unsigned tensor_d1,
                                                 unsigned stride_d0) {
  const unsigned lds_off = (unsigned)(unsigned long long)lds_dst;  // LDS byte offset
  const unsigned long long ga = (unsigned long long)gsrc;
  v4u g0;
  g0[0] = 1u;                                             // count=1, user-mode load
  g0[1] = lds_off;                                        // lds_addr (bytes)
  g0[2] = (unsigned)(ga & 0xffffffffu);                   // global_addr[31:0]
  g0[3] = (unsigned)((ga >> 32) & 0x01ffffffu) | (2u << 30);  // addr[56:32] | type=2
  v8i g1;
  g1[0] = (int)(1u << 16);                                // data_size=1 (2 bytes)
  g1[1] = (int)((tensor_d0 & 0xffffu) << 16);             // abar=0 | tensor_dim0 lo
  g1[2] = (int)((tensor_d0 >> 16) | ((tensor_d1 & 0xffffu) << 16));
  g1[3] = (int)((tensor_d1 >> 16) | ((tile_d0 & 0xffffu) << 16));
  g1[4] = (int)(tile_d1 & 0xffffu);                       // tile_dim1 | tile_dim2=0
  g1[5] = (int)stride_d0;                                 // tensor_dim0_stride[31:0]
  g1[6] = 0;                                              // stride[47:32]=0, d1s lo=0
  g1[7] = 0;
  v4i z4 = {};
#if defined(__clang_major__) && (__clang_major__ >= 23)
  v8i z8 = {};
  __builtin_amdgcn_tensor_load_to_lds(g0, g1, z4, z4, z8, 0);
#else
  __builtin_amdgcn_tensor_load_to_lds(g0, g1, z4, z4, 0);
#endif
}

// ---------------------------------------------------------------------------
// WMMA fragment helpers (CDNA5 ISA 7.12.2 layouts, wave32)
//  A 16x32 bf16: lane L: M = L&15 ; halves 0..7 = K (L>>4)*8.., halves 8..15 = K 16+(L>>4)*8..
//  B 32x16 bf16: lane L: N = L&15 ; halves 0..7 = K (L>>4)*8.., halves 8..15 = K 16+(L>>4)*8..
//  C/D 16x16 f32: lane L: N = L&15 ; VGPR j: M = j + 8*(L>>4)
// ---------------------------------------------------------------------------
__device__ __forceinline__ v16bf frag_a(const __bf16* rowp, int half) {
  v8bf lo = *(const v8bf*)(rowp + half * 8);
  v8bf hi = *(const v8bf*)(rowp + 16 + half * 8);
  return __builtin_shufflevector(lo, hi, 0,1,2,3,4,5,6,7,8,9,10,11,12,13,14,15);
}
__device__ __forceinline__ v16bf frag_b(const __bf16* colp, int half) {
  v8bf lo = *(const v8bf*)(colp + half * 8);
  v8bf hi = *(const v8bf*)(colp + 16 + half * 8);
  return __builtin_shufflevector(lo, hi, 0,1,2,3,4,5,6,7,8,9,10,11,12,13,14,15);
}
__device__ __forceinline__ v8f wmma_bf16(v16bf a, v16bf b, v8f c) {
  return __builtin_amdgcn_wmma_f32_16x16x32_bf16(false, a, false, b, (short)0, c,
                                                 false, false);
}

// ---------------------------------------------------------------------------
// GEMM: C = A(MTOTxD) @ W(DxD)^T  (y[r,c] = sum_k A[r,k]*W[c,k])
//   A_IS_F32: A is fp32 (converted to bf16 while staging to LDS).
//             A bf16 path stages the A tile with the Tensor Data Mover.
//   HEAD_OUT: write bf16 into [B,H,S,DK] layout; else fp32 dense [MTOT,D].
// Block tile 128x64, BK=32; 8 waves as 4(m) x 2(n), each wave 32x32 (2x2 frags)
// ---------------------------------------------------------------------------
#define BM 128
#define BN 64
#define BK 32

template<bool A_IS_F32, bool HEAD_OUT>
__global__ __launch_bounds__(256) void gemm_wt(const void* __restrict__ Av,
                                               const float* __restrict__ W,
                                               void* __restrict__ Cv) {
  __shared__ __bf16 As[BM][BK];   // 8 KB
  __shared__ __bf16 Bs[BN][BK];   // 4 KB
  const int tid  = threadIdx.x;
  const int wave = tid >> 5, lane = tid & 31;
  const int half = lane >> 4, lr = lane & 15;
  const int m0 = blockIdx.x * BM, n0 = blockIdx.y * BN;
  const int wm = wave >> 1, wn = wave & 1;

  v8f acc[2][2] = {};

  for (int k0 = 0; k0 < D_; k0 += BK) {
    // ---- stage A tile (128x32 bf16) ----
    if (A_IS_F32) {
      const int row = tid >> 1, seg = tid & 1;
      const float* ap = (const float*)Av + (size_t)(m0 + row) * D_ + k0 + seg * 16;
      if (k0 + BK < D_) __builtin_prefetch(ap + BK, 0, 0);
      #pragma unroll
      for (int t = 0; t < 4; ++t) {
        float4 f = ((const float4*)ap)[t];
        As[row][seg * 16 + t * 4 + 0] = (__bf16)f.x;
        As[row][seg * 16 + t * 4 + 1] = (__bf16)f.y;
        As[row][seg * 16 + t * 4 + 2] = (__bf16)f.z;
        As[row][seg * 16 + t * 4 + 3] = (__bf16)f.w;
      }
    } else {
      // TDM: 2D tile, tile 32x128 (elems x rows), row stride D_ elems
      if (wave == 0) {
        const __bf16* ap = (const __bf16*)Av + (size_t)m0 * D_ + k0;
        tdm_load_2d_bf16(&As[0][0], ap, BK, BM, D_, MTOT, D_);
      }
    }
    // ---- stage W tile as B (64 rows of W, 32 k) : 8 elems per thread ----
    {
      const int rw = tid >> 2, sw = tid & 3;
      const float* wp = W + (size_t)(n0 + rw) * D_ + k0 + sw * 8;
      if (k0 + BK < D_) __builtin_prefetch(wp + BK, 0, 0);
      float4 g0 = ((const float4*)wp)[0];
      float4 g1 = ((const float4*)wp)[1];
      Bs[rw][sw * 8 + 0] = (__bf16)g0.x; Bs[rw][sw * 8 + 1] = (__bf16)g0.y;
      Bs[rw][sw * 8 + 2] = (__bf16)g0.z; Bs[rw][sw * 8 + 3] = (__bf16)g0.w;
      Bs[rw][sw * 8 + 4] = (__bf16)g1.x; Bs[rw][sw * 8 + 5] = (__bf16)g1.y;
      Bs[rw][sw * 8 + 6] = (__bf16)g1.z; Bs[rw][sw * 8 + 7] = (__bf16)g1.w;
    }
    if (!A_IS_F32) {
      if (wave == 0) __builtin_amdgcn_s_wait_tensorcnt(0);  // TDM done before barrier
    }
    __syncthreads();

    v16bf af[2], bf[2];
    #pragma unroll
    for (int fm = 0; fm < 2; ++fm)
      af[fm] = frag_a(&As[wm * 32 + fm * 16 + lr][0], half);
    #pragma unroll
    for (int fn = 0; fn < 2; ++fn)
      bf[fn] = frag_b(&Bs[wn * 32 + fn * 16 + lr][0], half);
    #pragma unroll
    for (int fm = 0; fm < 2; ++fm)
      #pragma unroll
      for (int fn = 0; fn < 2; ++fn)
        acc[fm][fn] = wmma_bf16(af[fm], bf[fn], acc[fm][fn]);
    __syncthreads();
  }

  // ---- store ----
  #pragma unroll
  for (int fm = 0; fm < 2; ++fm) {
    #pragma unroll
    for (int fn = 0; fn < 2; ++fn) {
      const int rbase = m0 + wm * 32 + fm * 16 + 8 * half;
      const int col   = n0 + wn * 32 + fn * 16 + lr;
      #pragma unroll
      for (int j = 0; j < 8; ++j) {
        const int r = rbase + j;
        const float val = acc[fm][fn][j];
        if (HEAD_OUT) {
          const int b = r >> 11, s = r & (S_ - 1);
          const int h = col >> 6, dk = col & (DK_ - 1);
          ((__bf16*)Cv)[(((size_t)b * H_ + h) * S_ + s) * DK_ + dk] = (__bf16)val;
        } else {
          ((float*)Cv)[(size_t)r * D_ + col] = val;
        }
      }
    }
  }
}

// ---------------------------------------------------------------------------
// Attention: one block per (bh, 16-query-row tile). Full 16x2048 score block
// kept in LDS (128KB of the 320KB WGP LDS) -> single-pass softmax, attn
// written to HBM exactly once. PV: waves = 4 N-tiles x 2 K-halves.
// Dynamic LDS layout:
//   scores : 16*2048 f32            = 131072 B
//   qtile  : 16*64   bf16           =   2048 B
//   vt     : 2*64*32 bf16 (V^T)     =   8192 B
//   red    : 4*8*32  f32            =   4096 B     total 145408 B
// ---------------------------------------------------------------------------
#define ATTN_SMEM 145408

__global__ __launch_bounds__(256) void attn_kernel(
    const __bf16* __restrict__ Qh, const __bf16* __restrict__ Kh,
    const __bf16* __restrict__ Vh, float* __restrict__ attn,
    __bf16* __restrict__ concat) {
  extern __shared__ char smem_raw[];
  float*  scores = (float*)smem_raw;                       // [16][2048]
  __bf16* qtile  = (__bf16*)(smem_raw + 131072);           // [16][64]
  __bf16* vt     = (__bf16*)(smem_raw + 131072 + 2048);    // [2][64][32]
  float*  red    = (float*)(smem_raw + 131072 + 2048 + 8192); // [4][8][32]

  const int tid  = threadIdx.x;
  const int wave = tid >> 5, lane = tid & 31;
  const int half = lane >> 4, lr = lane & 15;
  const int bh    = blockIdx.y;          // b*H + h
  const int qbase = blockIdx.x * 16;

  // ---- stage Q tile (16x64 bf16, contiguous 2KB) via Tensor Data Mover ----
  if (wave == 0) {
    const __bf16* qp = Qh + ((size_t)bh * S_ + qbase) * DK_;
    tdm_load_2d_bf16(qtile, qp, /*tile_d0=*/16 * DK_, /*tile_d1=*/1,
                     /*tensor_d0=*/16 * DK_, /*tensor_d1=*/1, /*stride=*/16 * DK_);
    __builtin_amdgcn_s_wait_tensorcnt(0);
  }
  __syncthreads();

  // Q fragments: A = Q (16 x DK), two K=32 fragments
  const v16bf aq0 = frag_a(qtile + lr * DK_ + 0, half);
  const v16bf aq1 = frag_a(qtile + lr * DK_ + 32, half);

  // ---- QK^T: each wave owns key tiles jt = wave, wave+8, ... (16 tiles) ----
  for (int jt = wave; jt < S_ / 16; jt += 8) {
    const int k0 = jt * 16;
    const __bf16* kp = Kh + ((size_t)bh * S_ + k0 + lr) * DK_;   // key row (contig)
    if (jt + 8 < S_ / 16) __builtin_prefetch(kp + 8 * 16 * DK_, 0, 0);
    const v16bf bk0 = frag_b(kp, half);        // features 0..31
    const v16bf bk1 = frag_b(kp + 32, half);   // features 32..63
    v8f sc = {};
    sc = wmma_bf16(aq0, bk0, sc);
    sc = wmma_bf16(aq1, bk1, sc);
    #pragma unroll
    for (int j = 0; j < 8; ++j) {
      const int ql  = j + 8 * half;            // local query row
      const int key = k0 + lr;
      float v = sc[j] * 0.125f;                // 1/sqrt(64)
      if (key > qbase + ql) v = -__builtin_inff();   // causal mask
      scores[ql * S_ + key] = v;
    }
  }
  __syncthreads();

  // ---- softmax over each of the 16 rows (16 threads/row, half-wave shfl) ----
  {
    const int row = tid >> 4, chunk = tid & 15;   // 128 cols per thread
    float* srow = scores + row * S_ + chunk * 128;
    float mx = -3.4e38f;
    #pragma unroll
    for (int i = 0; i < 128; i += 4) {
      float4 f = *(float4*)(srow + i);
      mx = fmaxf(mx, fmaxf(fmaxf(f.x, f.y), fmaxf(f.z, f.w)));
    }
    #pragma unroll
    for (int off = 8; off; off >>= 1) mx = fmaxf(mx, __shfl_xor(mx, off, 16));
    float sum = 0.f;
    #pragma unroll
    for (int i = 0; i < 128; i += 4) {
      float4 f = *(float4*)(srow + i);
      f.x = __expf(f.x - mx); f.y = __expf(f.y - mx);
      f.z = __expf(f.z - mx); f.w = __expf(f.w - mx);
      sum += f.x + f.y + f.z + f.w;
      *(float4*)(srow + i) = f;
    }
    #pragma unroll
    for (int off = 8; off; off >>= 1) sum += __shfl_xor(sum, off, 16);
    const float inv = 1.f / sum;
    float* arow = attn + ((size_t)bh * S_ + qbase + row) * S_ + chunk * 128;
    #pragma unroll
    for (int i = 0; i < 128; i += 4) {
      float4 f = *(float4*)(srow + i);
      f.x *= inv; f.y *= inv; f.z *= inv; f.w *= inv;
      *(float4*)(srow + i) = f;        // keep normalized P in LDS for PV
      *(float4*)(arow + i) = f;        // materialize attn (the 512MB output)
    }
  }
  __syncthreads();

  // ---- PV: out(16x64) = P(16x2048) @ V(2048x64) ----
  const int ntile = wave & 3;          // dk tile (16 wide)
  const int khalf = wave >> 2;         // key half (0..1023 / 1024..2047)
  v8f oacc = {};
  for (int c = 0; c < 32; ++c) {       // 32-key chunks within each half
    // stage V^T for both halves: vt[hh][dk][kk]
    {
      const int r = tid >> 2;          // 0..63 (32 keys per half)
      const int hh = r >> 5, kk = r & 31, seg = tid & 3;
      const __bf16* vp =
          Vh + ((size_t)bh * S_ + hh * 1024 + c * 32 + kk) * DK_ + seg * 16;
      v8bf a0 = *(const v8bf*)vp;
      v8bf a1 = *(const v8bf*)(vp + 8);
      __bf16* dst = vt + hh * (64 * 32);
      #pragma unroll
      for (int i = 0; i < 8; ++i) dst[(seg * 16 + i) * 32 + kk] = a0[i];
      #pragma unroll
      for (int i = 0; i < 8; ++i) dst[(seg * 16 + 8 + i) * 32 + kk] = a1[i];
    }
    __syncthreads();

    // A fragment from normalized P (fp32 in LDS -> bf16)
    const int kb = khalf * 1024 + c * 32;
    const float* prow = scores + lr * S_ + kb;
    v16bf ap;
    #pragma unroll
    for (int i = 0; i < 8; ++i) ap[i]     = (__bf16)prow[half * 8 + i];
    #pragma unroll
    for (int i = 0; i < 8; ++i) ap[8 + i] = (__bf16)prow[16 + half * 8 + i];

    // B fragment from vt: N = dk, K = key (contiguous along kk)
    const __bf16* vrow = vt + khalf * (64 * 32) + (ntile * 16 + lr) * 32;
    v8bf lo = *(const v8bf*)(vrow + half * 8);
    v8bf hi = *(const v8bf*)(vrow + 16 + half * 8);
    v16bf bp = __builtin_shufflevector(lo, hi, 0,1,2,3,4,5,6,7,8,9,10,11,12,13,14,15);

    oacc = wmma_bf16(ap, bp, oacc);
    __syncthreads();
  }

  // ---- reduce the two K-halves, write concat [B,S,D] bf16 ----
  if (khalf == 1) {
    #pragma unroll
    for (int j = 0; j < 8; ++j) red[(ntile * 8 + j) * 32 + lane] = oacc[j];
  }
  __syncthreads();
  if (khalf == 0) {
    const int b = bh >> 4, h = bh & 15;
    #pragma unroll
    for (int j = 0; j < 8; ++j) {
      const float v = oacc[j] + red[(ntile * 8 + j) * 32 + lane];
      const int ql = j + 8 * half;
      const int dk = ntile * 16 + lr;
      concat[((size_t)b * S_ + qbase + ql) * D_ + h * DK_ + dk] = (__bf16)v;
    }
  }
}

// ---------------------------------------------------------------------------
extern "C" void kernel_launch(void* const* d_in, const int* in_sizes, int n_in,
                              void* d_out, int out_size, void* d_ws, size_t ws_size,
                              hipStream_t stream) {
  const float* q  = (const float*)d_in[0];
  const float* k  = (const float*)d_in[1];
  const float* v  = (const float*)d_in[2];
  // d_in[3] = causal mask (int32) -- derived analytically in the kernel
  const float* Wq = (const float*)d_in[4];
  const float* Wk = (const float*)d_in[5];
  const float* Wv = (const float*)d_in[6];
  const float* Wo = (const float*)d_in[7];

  const size_t HE = (size_t)B_ * H_ * S_ * DK_;   // 4,194,304 elems
  __bf16* Qh = (__bf16*)d_ws;
  __bf16* Kh = Qh + HE;
  __bf16* Vh = Kh + HE;
  __bf16* Cc = Vh + HE;                           // concat, bf16 [B,S,D]

  float* out  = (float*)d_out;                    // [B,S,D] fp32
  float* attn = out + (size_t)B_ * S_ * D_;       // [B,H,S,S] fp32

  dim3 ggrid(MTOT / BM, D_ / BN);                 // 32 x 16
  gemm_wt<true,  true><<<ggrid, 256, 0, stream>>>(q, Wq, Qh);
  gemm_wt<true,  true><<<ggrid, 256, 0, stream>>>(k, Wk, Kh);
  gemm_wt<true,  true><<<ggrid, 256, 0, stream>>>(v, Wv, Vh);

  (void)hipFuncSetAttribute((const void*)attn_kernel,
                            hipFuncAttributeMaxDynamicSharedMemorySize, ATTN_SMEM);
  attn_kernel<<<dim3(S_ / 16, B_ * H_), 256, ATTN_SMEM, stream>>>(Qh, Kh, Vh,
                                                                  attn, Cc);

  gemm_wt<false, false><<<ggrid, 256, 0, stream>>>(Cc, Wo, out);
}
